// NGPImage_51101520888560
// MI455X (gfx1250) — matrix operand order repaired and checked by reading, not compiled, not run on hardware
//
#include <hip/hip_runtime.h>
#include <hip/hip_fp16.h>

// Instant-NGP 2D hash-grid encode + 3-layer MLP, fused, for gfx1250 (MI455X).
// Phase 1: per-thread hash gather (L2-resident 128MB tables) -> f16 LDS.
// Phase 2: per-wave WMMA GEMMs (v_wmma_f32_16x16x32_f16) through LDS.

#define NGP_L 16
#define NGP_T (1u << 20)          // power of two -> hash mod is a mask
#define NGP_TMASK (NGP_T - 1u)
#define NGP_PRIME 2654435761u

typedef _Float16 h8   __attribute__((ext_vector_type(8)));
typedef _Float16 v16h __attribute__((ext_vector_type(16)));
typedef float    v8f  __attribute__((ext_vector_type(8)));

// floor(16 * 64^(l/15)) for l = 0..15 (instant-NGP geometric schedule)
__constant__ float kRes[NGP_L] = {
    16.f, 21.f, 27.f, 36.f, 48.f, 64.f, 84.f, 111.f,
    147.f, 194.f, 256.f, 337.f, 445.f, 588.f, 776.f, 1024.f};

__device__ __forceinline__ v8f wmma_f16(v16h a, v16h b, v8f c) {
    return __builtin_amdgcn_wmma_f32_16x16x32_f16(
        /*neg_a=*/false, a, /*neg_b=*/false, b,
        /*c_mod=*/(short)0, c, /*reuse_a=*/false, /*reuse_b=*/false);
}

// A fragment (16x32 f16) from row-major LDS act[rows][64].
// Layout: halves 0-15 hold K = 16*(lane>=16)+0..15 (matches B frag),
// M = lane%16.  One contiguous 32B run -> two ds_load_b128.
__device__ __forceinline__ v16h load_a_frag(const _Float16* act, int rowBase,
                                            int lane, int colBase) {
    int m  = lane & 15;
    int hi = lane >> 4;
    const _Float16* p = act + (rowBase + m) * 64 + colBase + hi * 16;
    h8 lo = *(const h8*)p;
    h8 hh = *(const h8*)(p + 8);
    return __builtin_shufflevector(lo, hh, 0, 1, 2, 3, 4, 5, 6, 7,
                                   8, 9, 10, 11, 12, 13, 14, 15);
}

// B fragment (32x16 f16), pre-swizzled in LDS as [lane][16 halves] -> contiguous loads.
__device__ __forceinline__ v16h load_b_frag(const _Float16* frag, int lane) {
    const _Float16* p = frag + lane * 16;
    h8 lo = *(const h8*)p;
    h8 hh = *(const h8*)(p + 8);
    return __builtin_shufflevector(lo, hh, 0, 1, 2, 3, 4, 5, 6, 7,
                                   8, 9, 10, 11, 12, 13, 14, 15);
}

// Store relu(D) (16x16 f32 tile) back to LDS act rows as f16.
// D layout: N = lane%16, M = r + 8*(lane>=16).
__device__ __forceinline__ void store_relu(_Float16* act, int rowBase, int lane,
                                           int colBase, v8f c) {
    int n  = lane & 15;
    int hi = lane >> 4;
#pragma unroll
    for (int r = 0; r < 8; ++r) {
        float v = fmaxf(c[r], 0.0f);
        act[(rowBase + r + 8 * hi) * 64 + colBase + n] = (_Float16)v;
    }
}

__global__ __launch_bounds__(256) void ngp_fused_kernel(
    const float* __restrict__ x,       // [N,2]
    const float* __restrict__ tables,  // [L,T,2]
    const float* __restrict__ w0,      // [32,64]
    const float* __restrict__ b0,      // [64]
    const float* __restrict__ w1,      // [64,64]
    const float* __restrict__ b1,      // [64]
    const float* __restrict__ w2,      // [64,4]
    const float* __restrict__ b2,      // [4]
    float* __restrict__ out,           // [N,4]
    int n) {
    // act: 256 rows x 64 f16 cols (cols 0..31 = encoding, later overwritten by h1/h2)
    __shared__ __align__(16) _Float16 act[256 * 64];                 // 32 KB
    __shared__ __align__(16) _Float16 w0s[4 * 512];                  // 4 KB  (1 kt x 4 ct)
    __shared__ __align__(16) _Float16 w1s[8 * 512];                  // 8 KB  (2 kt x 4 ct)
    __shared__ __align__(16) _Float16 w2s[2 * 512];                  // 2 KB  (2 kt x 1 ct, N padded to 16)
    __shared__ float b0s[64], b1s[64], b2s[16];

    const int tid  = threadIdx.x;
    const int base = blockIdx.x * 256;

    // ---- stage weights into pre-swizzled f16 fragments --------------------
    // B frag element (k%32, n%16) lives at lane = n%16 + 16*((k%32)>=16), half = k%16.
    for (int e = tid; e < 2 * 512; e += 256) w2s[e] = (_Float16)0.0f;  // zero-pad w2
    if (tid < 16) b2s[tid] = 0.0f;

    for (int e = tid; e < 32 * 64; e += 256) {
        int k = e >> 6, nn = e & 63;
        int ct = nn >> 4, nc = nn & 15;
        int lane = nc + (k & 16);
        w0s[ct * 512 + lane * 16 + (k & 15)] = (_Float16)w0[e];
    }
    for (int e = tid; e < 64 * 64; e += 256) {
        int k = e >> 6, nn = e & 63;
        int kt = k >> 5, kk = k & 31;
        int ct = nn >> 4, nc = nn & 15;
        int lane = nc + (kk & 16);
        w1s[(kt * 4 + ct) * 512 + lane * 16 + (kk & 15)] = (_Float16)w1[e];
    }
    for (int e = tid; e < 64 * 4; e += 256) {
        int k = e >> 2, nn = e & 3;
        int kt = k >> 5, kk = k & 31;
        int lane = nn + (kk & 16);
        w2s[kt * 512 + lane * 16 + (kk & 15)] = (_Float16)w2[e];
    }
    if (tid < 64) b0s[tid] = b0[tid];
    if (tid < 64) b1s[tid] = b1[tid];
    if (tid < 4)  b2s[tid] = b2[tid];

    // ---- phase 1: hash-grid encode (one point per thread) -----------------
    int i  = base + tid;
    int ii = (i < n) ? i : (n - 1);  // clamp, keep EXEC full
    const float2 xy = ((const float2*)x)[ii];
#pragma unroll
    for (int l = 0; l < NGP_L; ++l) {
        float r  = kRes[l];
        float s0 = xy.x * r, s1 = xy.y * r;
        float fl0 = floorf(s0), fl1 = floorf(s1);
        float fr0 = s0 - fl0, fr1 = s1 - fl1;
        unsigned p0 = (unsigned)fl0, p1 = (unsigned)fl1;
        const float2* tb = (const float2*)tables + (size_t)l * NGP_T;
        unsigned hy0 = p1 * NGP_PRIME;
        unsigned hy1 = (p1 + 1u) * NGP_PRIME;
        unsigned i00 = (p0 ^ hy0) & NGP_TMASK;
        unsigned i01 = (p0 ^ hy1) & NGP_TMASK;
        unsigned i10 = ((p0 + 1u) ^ hy0) & NGP_TMASK;
        unsigned i11 = ((p0 + 1u) ^ hy1) & NGP_TMASK;
        float2 f00 = tb[i00];
        float2 f01 = tb[i01];
        float2 f10 = tb[i10];
        float2 f11 = tb[i11];
        float w00 = (1.0f - fr0) * (1.0f - fr1);
        float w01 = (1.0f - fr0) * fr1;
        float w10 = fr0 * (1.0f - fr1);
        float w11 = fr0 * fr1;
        float e0 = f00.x * w00 + f01.x * w01 + f10.x * w10 + f11.x * w11;
        float e1 = f00.y * w00 + f01.y * w01 + f10.y * w10 + f11.y * w11;
        act[tid * 64 + 2 * l + 0] = (_Float16)e0;
        act[tid * 64 + 2 * l + 1] = (_Float16)e1;
    }
    __syncthreads();

    // ---- phase 2: MLP via WMMA (wave-level; each wave owns 2 row tiles) ----
    const int lane = tid & 31;
    const int wave = tid >> 5;

#pragma unroll
    for (int t = 0; t < 2; ++t) {
        const int rowBase = (wave * 2 + t) * 16;

        // layer 0: [16x32] x [32x64] -> relu -> act cols 0..63
        {
            v16h a0 = load_a_frag(act, rowBase, lane, 0);
#pragma unroll
            for (int ct = 0; ct < 4; ++ct) {
                float bb = b0s[ct * 16 + (lane & 15)];
                v8f c;
#pragma unroll
                for (int r = 0; r < 8; ++r) c[r] = bb;
                v16h bf = load_b_frag(w0s + ct * 512, lane);
                c = wmma_f16(a0, bf, c);
                store_relu(act, rowBase, lane, ct * 16, c);
            }
        }

        // layer 1: [16x64] x [64x64] -> relu -> act cols 0..63 (in place)
        {
            v16h a0 = load_a_frag(act, rowBase, lane, 0);
            v16h a1 = load_a_frag(act, rowBase, lane, 32);
#pragma unroll
            for (int ct = 0; ct < 4; ++ct) {
                float bb = b1s[ct * 16 + (lane & 15)];
                v8f c;
#pragma unroll
                for (int r = 0; r < 8; ++r) c[r] = bb;
                c = wmma_f16(a0, load_b_frag(w1s + ct * 512, lane), c);
                c = wmma_f16(a1, load_b_frag(w1s + (4 + ct) * 512, lane), c);
                store_relu(act, rowBase, lane, ct * 16, c);
            }
        }

        // layer 2: [16x64] x [64x16(pad)] -> sigmoid -> out[:,0:4]
        {
            v16h a0 = load_a_frag(act, rowBase, lane, 0);
            v16h a1 = load_a_frag(act, rowBase, lane, 32);
            float bb = b2s[lane & 15];
            v8f c;
#pragma unroll
            for (int r = 0; r < 8; ++r) c[r] = bb;
            c = wmma_f16(a0, load_b_frag(w2s + 0, lane), c);
            c = wmma_f16(a1, load_b_frag(w2s + 512, lane), c);

            int nc = lane & 15;
            int hi = lane >> 4;
            if (nc < 4) {
#pragma unroll
                for (int r = 0; r < 8; ++r) {
                    int pt = base + rowBase + r + 8 * hi;
                    if (pt < n) {
                        float v = 1.0f / (1.0f + __expf(-c[r]));
                        out[pt * 4 + nc] = v;
                    }
                }
            }
        }
    }
}

extern "C" void kernel_launch(void* const* d_in, const int* in_sizes, int n_in,
                              void* d_out, int out_size, void* d_ws, size_t ws_size,
                              hipStream_t stream) {
    (void)n_in; (void)d_ws; (void)ws_size; (void)out_size;
    const float* x      = (const float*)d_in[0];
    const float* tables = (const float*)d_in[1];
    const float* w0     = (const float*)d_in[2];
    const float* b0     = (const float*)d_in[3];
    const float* w1     = (const float*)d_in[4];
    const float* b1     = (const float*)d_in[5];
    const float* w2     = (const float*)d_in[6];
    const float* b2     = (const float*)d_in[7];
    float* out          = (float*)d_out;

    const int n = in_sizes[0] / 2;           // x is [N,2]
    const int blocks = (n + 255) / 256;      // 256 points per block (8 wave32s)
    ngp_fused_kernel<<<blocks, 256, 0, stream>>>(x, tables, w0, b0, w1, b1,
                                                 w2, b2, out, n);
}
